// Decoder_82970178224185
// MI455X (gfx1250) — matrix operand order repaired and checked by reading, not compiled, not run on hardware
//
#include <hip/hip_runtime.h>

typedef __bf16 bf16;
typedef __attribute__((ext_vector_type(16))) __bf16 v16bf;
typedef __attribute__((ext_vector_type(8)))  __bf16 v8bf;
typedef __attribute__((ext_vector_type(4)))  __bf16 v4bf;
typedef __attribute__((ext_vector_type(8)))  float  v8f;
typedef __attribute__((ext_vector_type(4)))  unsigned int v4u;
typedef __attribute__((ext_vector_type(8)))  int v8i;
typedef __attribute__((ext_vector_type(4)))  int v4i;

// Problem constants (from reference)
#define BQ   8
#define LCB  1024
#define LPB  512
#define DB   512
#define HB   8
#define HDB  64
#define FB   2048
#define VB   70
#define NLB  4

static __device__ __forceinline__ v16bf concat8(v8bf a, v8bf b) {
  return __builtin_shufflevector(a, b, 0, 1, 2, 3, 4, 5, 6, 7, 8, 9, 10, 11, 12, 13, 14, 15);
}

// ---- DPP16 cross-lane reductions over each 16-lane half (no LDS traffic) ---
template <int CTRL>
static __device__ __forceinline__ float dpp_mov(float v) {
  return __int_as_float(__builtin_amdgcn_update_dpp(
      __float_as_int(v), __float_as_int(v), CTRL, 0xF, 0xF, false));
}
static __device__ __forceinline__ float red_max16(float v) {
  v = fmaxf(v, dpp_mov<0xB1>(v));    // quad_perm [1,0,3,2]
  v = fmaxf(v, dpp_mov<0x4E>(v));    // quad_perm [2,3,0,1]
  v = fmaxf(v, dpp_mov<0x141>(v));   // row_half_mirror
  v = fmaxf(v, dpp_mov<0x140>(v));   // row_mirror
  return v;
}
static __device__ __forceinline__ float red_sum16(float v) {
  v += dpp_mov<0xB1>(v);
  v += dpp_mov<0x4E>(v);
  v += dpp_mov<0x141>(v);
  v += dpp_mov<0x140>(v);
  return v;
}

// ---- Tensor Data Mover: 2D bf16 tile DMA global -> LDS (ISA ch.8 D#) ------
// Descriptor groups per CDNA5 ISA 8.3/8.4; 2-D tensor: groups 2,3 zero.
// 6-arg builtin form (clang-23 / therock-10.0 headers).
static __device__ __forceinline__ void tdm_load_2d_bf16(
    const bf16* gaddr, unsigned lds_off, unsigned tile_x, unsigned tile_y,
    unsigned tensor_x, unsigned tensor_y, unsigned stride_x) {
  const unsigned long long ga = (unsigned long long)gaddr;
  v4u g0;
  g0[0] = 1u;                                        // count=1 (valid, user)
  g0[1] = lds_off;                                   // LDS byte address
  g0[2] = (unsigned)(ga & 0xFFFFFFFFu);              // global_addr[31:0]
  g0[3] = (unsigned)((ga >> 32) & 0x01FFFFFFu) | (2u << 30);  // [56:32] | type=2
  v8i g1;
  g1[0] = (int)(1u << 16);                           // data_size=1 (2B), mask=0
  g1[1] = (int)((tensor_x & 0xFFFFu) << 16);         // tensor_dim0[15:0]
  g1[2] = (int)(((tensor_x >> 16) & 0xFFFFu) | ((tensor_y & 0xFFFFu) << 16));
  g1[3] = (int)(((tensor_y >> 16) & 0xFFFFu) | ((tile_x & 0xFFFFu) << 16));
  g1[4] = (int)(tile_y & 0xFFFFu);                   // tile_dim1; tile_dim2=0
  g1[5] = (int)stride_x;                             // tensor_dim0_stride[31:0]
  g1[6] = 0;                                         // stride hi / dim1_stride lo
  g1[7] = 0;
  const v4i z4 = {0, 0, 0, 0};
  const v8i z8 = {0, 0, 0, 0, 0, 0, 0, 0};
  __builtin_amdgcn_tensor_load_to_lds(g0, g1, z4, z4, z8, 0);
}

// ---------------------------------------------------------------------------
// f32 -> bf16 conversion, vectorized (n multiple of 4; all our tensors are)
// ---------------------------------------------------------------------------
__global__ void f2bf_kernel(const float* __restrict__ in, bf16* __restrict__ out, int n4) {
  const int i = blockIdx.x * blockDim.x + threadIdx.x;
  if (i < n4) {
    const float4 f = ((const float4*)in)[i];
    v4bf o;
    o[0] = (bf16)f.x; o[1] = (bf16)f.y; o[2] = (bf16)f.z; o[3] = (bf16)f.w;
    ((v4bf*)out)[i] = o;
  }
}

// ---------------------------------------------------------------------------
// Embedding: x = tok_emb[cds]*sqrt(D) + pos_emb[l]; also bf16 copy
// ---------------------------------------------------------------------------
__global__ __launch_bounds__(128) void embed_kernel(
    const int* __restrict__ cds, const float* __restrict__ tok_emb,
    const float* __restrict__ pos_emb, float* __restrict__ x, bf16* __restrict__ xb) {
  const int row = blockIdx.x;            // b*LC + l
  const int l   = row & (LCB - 1);
  const int tok = cds[row];
  const int d   = threadIdx.x * 4;
  const float sc = 22.62741699796952f;   // sqrt(512)
  const float4 te = *(const float4*)&tok_emb[tok * DB + d];
  const float4 pe = *(const float4*)&pos_emb[l * DB + d];
  float4 v;
  v.x = te.x * sc + pe.x; v.y = te.y * sc + pe.y;
  v.z = te.z * sc + pe.z; v.w = te.w * sc + pe.w;
  *(float4*)&x[(size_t)row * DB + d] = v;
  v4bf o; o[0] = (bf16)v.x; o[1] = (bf16)v.y; o[2] = (bf16)v.z; o[3] = (bf16)v.w;
  *(v4bf*)&xb[(size_t)row * DB + d] = o;
}

// ---------------------------------------------------------------------------
// bf16 WMMA GEMM: C[M,N] = A[M,K] @ W[K,N] + bias ; optional ReLU, f32/bf16
// outputs. M % 128 == 0, K % 64 == 0. NBOUND=true handles arbitrary N.
// Block 256 (8 waves); block tile 128x128; wave tile 32x64 (8 accumulators).
// A tile staged by the TENSOR DATA MOVER (tensor_load_to_lds) issued by wave
// 0, overlapped with the B tile's transposed VGPR staging; B-fragments are
// then contiguous b128 LDS loads.
// ---------------------------------------------------------------------------
template <bool NBOUND>
__global__ __launch_bounds__(256) void gemm_bf16_kernel(
    const bf16* __restrict__ A, const bf16* __restrict__ W,
    const float* __restrict__ bias, float* __restrict__ outF,
    bf16* __restrict__ outB, int M, int N, int K, int relu) {
  __shared__ __align__(16) bf16 As[128][64];    // [m][k], row = 128B (TDM dest)
  __shared__ __align__(16) bf16 Bst[128][80];   // transposed [n][k], row = 160B

  const int t = threadIdx.x, lane = t & 31, wid = t >> 5;
  const int m0 = blockIdx.y * 128, n0 = blockIdx.x * 128;
  const int wm = wid & 3;               // wave M sub-tile: 32 rows
  const int wn = wid >> 2;              // wave N sub-tile: 64 cols
  const int n16 = lane & 15, hk = lane >> 4;

  v8f acc[2][4] = {};

  const int bk = t >> 2, bn = (t & 3) * 32;  // B coop: 64 k x 128 n
  const unsigned asLds = (unsigned)(unsigned long long)&As[0][0]; // LDS offset

  for (int k0 = 0; k0 < K; k0 += 64) {
    __syncthreads();
    if (wid == 0) {  // TDM: DMA the 128x64 bf16 A tile into LDS (async)
      tdm_load_2d_bf16(&A[(size_t)m0 * K + k0], asLds,
                       /*tile_x=*/64, /*tile_y=*/128,
                       /*tensor_x=*/(unsigned)K, /*tensor_y=*/(unsigned)M,
                       /*stride_x=*/(unsigned)K);
    }
    { // B tile: b128 global loads, transposed scatter into LDS (imm offsets)
      const int gn = n0 + bn;
      bf16 tv[32];
      if (!NBOUND || (gn + 32 <= N)) {
        const bf16* gw = &W[(size_t)(k0 + bk) * N + gn];
#pragma unroll
        for (int c = 0; c < 4; ++c)
          *(uint4*)&tv[c * 8] = *(const uint4*)(gw + c * 8);
      } else {
#pragma unroll
        for (int j = 0; j < 32; ++j)
          tv[j] = (gn + j < N) ? W[(size_t)(k0 + bk) * N + gn + j] : (bf16)0.f;
      }
      bf16* bp = &Bst[bn][bk];
#pragma unroll
      for (int j = 0; j < 32; ++j) bp[j * 80] = tv[j];
    }
    if (wid == 0) __builtin_amdgcn_s_wait_tensorcnt(0);  // A tile landed
    __syncthreads();

#pragma unroll
    for (int kc = 0; kc < 64; kc += 32) {
      v16bf af[2], bfr[4];
#pragma unroll
      for (int am = 0; am < 2; ++am) {
        const bf16* ap = &As[wm * 32 + am * 16 + n16][kc + hk * 8];
        af[am] = concat8(*(const v8bf*)ap, *(const v8bf*)(ap + 16));
      }
#pragma unroll
      for (int bq = 0; bq < 4; ++bq) {
        const bf16* bp = &Bst[wn * 64 + bq * 16 + n16][kc + hk * 8];
        bfr[bq] = concat8(*(const v8bf*)bp, *(const v8bf*)(bp + 16));
      }
#pragma unroll
      for (int am = 0; am < 2; ++am)
#pragma unroll
        for (int bq = 0; bq < 4; ++bq)
          acc[am][bq] = __builtin_amdgcn_wmma_f32_16x16x32_bf16(
              false, af[am], false, bfr[bq], (short)0, acc[am][bq], false, false);
    }
  }

#pragma unroll
  for (int am = 0; am < 2; ++am) {
#pragma unroll
    for (int bq = 0; bq < 4; ++bq) {
      const int gn = n0 + wn * 64 + bq * 16 + n16;
      if (NBOUND && gn >= N) continue;
      const float bv = bias ? bias[gn] : 0.f;
#pragma unroll
      for (int r = 0; r < 8; ++r) {
        const int gm = m0 + wm * 32 + am * 16 + r + 8 * hk;
        float v = acc[am][bq][r] + bv;
        if (relu) v = fmaxf(v, 0.f);
        if (outF) outF[(size_t)gm * N + gn] = v;
        if (outB) outB[(size_t)gm * N + gn] = (bf16)v;
      }
    }
  }
}

// ---------------------------------------------------------------------------
// Flash attention (WMMA): block = 64 query rows of one (b,h); 4 waves, each
// wave owns 16 rows; streams 32-key tiles with online softmax (DPP16
// reductions). V tile staged TRANSPOSED in LDS so V-fragments are b128 loads.
// ---------------------------------------------------------------------------
__global__ __launch_bounds__(128) void flash_attn_kernel(
    const bf16* __restrict__ Q, const bf16* __restrict__ Km,
    const bf16* __restrict__ Vm, bf16* __restrict__ Octx,
    int Lq, int Lk, int causal) {
  __shared__ __align__(16) bf16 Vst[64][40];   // transposed [d][key], row = 80B
  __shared__ __align__(16) bf16 Pb[4][16][32]; // per-wave P tile [row][key]

  const int t = threadIdx.x, lane = t & 31, wid = t >> 5;
  const int bh = blockIdx.y, b = bh / HB, h = bh % HB;
  const int q0 = blockIdx.x * 64;
  const int rowbase = q0 + wid * 16;
  const int n16 = lane & 15, hk = lane >> 4;

  // Q fragments for the two 32-wide d-chunks (d = 64)
  v16bf qf0, qf1;
  {
    const int qr = rowbase + n16;
    const bf16* qp0 = Q + ((size_t)(b * Lq + qr)) * DB + h * HDB + hk * 8;
    qf0 = concat8(*(const v8bf*)qp0, *(const v8bf*)(qp0 + 16));
    qf1 = concat8(*(const v8bf*)(qp0 + 32), *(const v8bf*)(qp0 + 48));
  }

  v8f o0 = {}, o1 = {}, o2 = {}, o3 = {};
  float mrow[8], lrow[8];
#pragma unroll
  for (int r = 0; r < 8; ++r) { mrow[r] = -1e30f; lrow[r] = 0.f; }

  const int ktEnd = causal ? ((Lk < q0 + 64) ? Lk : (q0 + 64)) : Lk;
  for (int key0 = 0; key0 < ktEnd; key0 += 32) {
    __syncthreads();
    { // cooperative V tile (32 keys x 64 d) -> transposed LDS
      const int vk = t >> 2, dc = (t & 3) * 16;
      const bf16* vp = Vm + ((size_t)(b * Lk + key0 + vk)) * DB + h * HDB + dc;
      bf16 tv[16];
      *(uint4*)&tv[0] = *(const uint4*)vp;
      *(uint4*)&tv[8] = *(const uint4*)(vp + 8);
      bf16* dst = &Vst[dc][vk];
#pragma unroll
      for (int j = 0; j < 16; ++j) dst[j * 40] = tv[j];
    }
    __syncthreads();

    // scores: two 16-key sub-tiles, each = 2 WMMAs over d (K-fragments are
    // contiguous b128 loads straight from global)
    v8f s0 = {}, s1 = {};
#pragma unroll
    for (int sub = 0; sub < 2; ++sub) {
      const int kcol = key0 + sub * 16 + n16;   // lane = key column
      const bf16* kp = Km + ((size_t)(b * Lk + kcol)) * DB + h * HDB + hk * 8;
      const v16bf bk0 = concat8(*(const v8bf*)kp, *(const v8bf*)(kp + 16));
      const v16bf bk1 = concat8(*(const v8bf*)(kp + 32), *(const v8bf*)(kp + 48));
      v8f s = {};
      s = __builtin_amdgcn_wmma_f32_16x16x32_bf16(false, qf0, false, bk0, (short)0, s, false, false);
      s = __builtin_amdgcn_wmma_f32_16x16x32_bf16(false, qf1, false, bk1, (short)0, s, false, false);
      if (sub == 0) s0 = s; else s1 = s;
    }

    const int col0 = key0 + n16, col1 = key0 + 16 + n16;
#pragma unroll
    for (int r = 0; r < 8; ++r) {
      const int gr = rowbase + r + 8 * hk;
      float a = s0[r] * 0.125f;                 // 1/sqrt(64)
      float c = s1[r] * 0.125f;
      if (causal) { if (col0 > gr) a = -1e30f; if (col1 > gr) c = -1e30f; }
      s0[r] = a; s1[r] = c;
    }

    // online softmax with DPP16 reductions; stage P in per-wave LDS to
    // re-layout C-frag -> A-frag
#pragma unroll
    for (int r = 0; r < 8; ++r) {
      const float tm = red_max16(fmaxf(s0[r], s1[r]));
      const float mn = fmaxf(mrow[r], tm);
      const float scale = __expf(mrow[r] - mn);
      const float p0 = __expf(s0[r] - mn);
      const float p1 = __expf(s1[r] - mn);
      const float rs = red_sum16(p0 + p1);
      lrow[r] = lrow[r] * scale + rs;
      mrow[r] = mn;
      o0[r] *= scale; o1[r] *= scale; o2[r] *= scale; o3[r] *= scale;
      Pb[wid][r + 8 * hk][n16]      = (bf16)p0;
      Pb[wid][r + 8 * hk][16 + n16] = (bf16)p1;
    }

    // P as A-fragment (16 rows x 32 keys): contiguous b128 loads
    const bf16* pp = &Pb[wid][n16][hk * 8];
    const v16bf pa = concat8(*(const v8bf*)pp, *(const v8bf*)(pp + 16));

#pragma unroll
    for (int dt = 0; dt < 4; ++dt) {
      const bf16* vp2 = &Vst[dt * 16 + n16][hk * 8];
      const v16bf vbf = concat8(*(const v8bf*)vp2, *(const v8bf*)(vp2 + 16));
      v8f* op = (dt == 0) ? &o0 : (dt == 1) ? &o1 : (dt == 2) ? &o2 : &o3;
      *op = __builtin_amdgcn_wmma_f32_16x16x32_bf16(false, pa, false, vbf, (short)0, *op, false, false);
    }
  }

#pragma unroll
  for (int r = 0; r < 8; ++r) {
    const int gr = rowbase + r + 8 * hk;
    const float inv = 1.0f / lrow[r];
    bf16* op = Octx + ((size_t)(b * Lq + gr)) * DB + h * HDB + n16;
    op[0]  = (bf16)(o0[r] * inv);
    op[16] = (bf16)(o1[r] * inv);
    op[32] = (bf16)(o2[r] * inv);
    op[48] = (bf16)(o3[r] * inv);
  }
}

// ---------------------------------------------------------------------------
// Residual add + LayerNorm over D=512; writes f32 state + bf16 copy
// ---------------------------------------------------------------------------
__global__ __launch_bounds__(256) void add_ln_kernel(
    float* __restrict__ x, const float* __restrict__ resid,
    const float* __restrict__ g, const float* __restrict__ bta,
    bf16* __restrict__ xb) {
  __shared__ float red[256];
  const int row = blockIdx.x, t = threadIdx.x;
  const size_t base = (size_t)row * DB;
  float v0 = x[base + t]       + resid[base + t];
  float v1 = x[base + t + 256] + resid[base + t + 256];
  red[t] = v0 + v1;
  __syncthreads();
  for (int s2 = 128; s2 > 0; s2 >>= 1) { if (t < s2) red[t] += red[t + s2]; __syncthreads(); }
  const float mu = red[0] * (1.0f / DB);
  __syncthreads();
  const float e0 = v0 - mu, e1 = v1 - mu;
  red[t] = e0 * e0 + e1 * e1;
  __syncthreads();
  for (int s2 = 128; s2 > 0; s2 >>= 1) { if (t < s2) red[t] += red[t + s2]; __syncthreads(); }
  const float rs = rsqrtf(red[0] * (1.0f / DB) + 1e-5f);
  const float y0 = e0 * rs * g[t]       + bta[t];
  const float y1 = e1 * rs * g[t + 256] + bta[t + 256];
  x[base + t] = y0;        x[base + t + 256] = y1;
  xb[base + t] = (bf16)y0; xb[base + t + 256] = (bf16)y1;
}

// ---------------------------------------------------------------------------
// Last-layer cross-attention probabilities -> d_out attn region [B,H,LC,LP]
// ---------------------------------------------------------------------------
__global__ __launch_bounds__(64) void ca_probs_kernel(
    const bf16* __restrict__ Qb, const bf16* __restrict__ Kb,
    float* __restrict__ attn) {
  __shared__ float sc[LPB];
  __shared__ float qsh[HDB];
  __shared__ float red[64];
  const int t = threadIdx.x, q = blockIdx.x;
  const int bh = blockIdx.y, b = bh / HB, h = bh % HB;
  const bf16* qp = Qb + ((size_t)(b * LCB + q)) * DB + h * HDB;
  if (t < HDB) qsh[t] = (float)qp[t];
  __syncthreads();
  for (int key = t; key < LPB; key += 64) {
    const bf16* kp = Kb + ((size_t)(b * LPB + key)) * DB + h * HDB;
    float dot = 0.f;
    for (int d = 0; d < HDB; ++d) dot += qsh[d] * (float)kp[d];
    sc[key] = dot * 0.125f;
  }
  __syncthreads();
  float mx = -1e30f;
  for (int key = t; key < LPB; key += 64) mx = fmaxf(mx, sc[key]);
  red[t] = mx; __syncthreads();
  for (int s2 = 32; s2 > 0; s2 >>= 1) { if (t < s2) red[t] = fmaxf(red[t], red[t + s2]); __syncthreads(); }
  mx = red[0]; __syncthreads();
  float sm = 0.f;
  for (int key = t; key < LPB; key += 64) { float e = __expf(sc[key] - mx); sc[key] = e; sm += e; }
  red[t] = sm; __syncthreads();
  for (int s2 = 32; s2 > 0; s2 >>= 1) { if (t < s2) red[t] += red[t + s2]; __syncthreads(); }
  const float inv = 1.f / red[0];
  float* ap = attn + (((size_t)(b * HB + h) * LCB + q) * LPB);
  for (int key = t; key < LPB; key += 64) ap[key] = sc[key] * inv;
}

// ---------------------------------------------------------------------------
// Host orchestration
// ---------------------------------------------------------------------------
extern "C" void kernel_launch(void* const* d_in, const int* in_sizes, int n_in,
                              void* d_out, int out_size, void* d_ws, size_t ws_size,
                              hipStream_t stream) {
  (void)in_sizes; (void)n_in; (void)out_size; (void)ws_size;
  const int Mq = BQ * LCB;   // 8192 decoder tokens
  const int Mk = BQ * LPB;   // 4096 protein tokens

  const int*   cds     = (const int*)d_in[0];
  const float* enc     = (const float*)d_in[1];
  const float* tok_emb = (const float*)d_in[4];
  const float* pos_emb = (const float*)d_in[5];
  const float *saw[4], *sab[4], *caw[4], *cab[4];
  for (int m = 0; m < 4; ++m) {
    saw[m] = (const float*)d_in[6 + 2 * m];  sab[m] = (const float*)d_in[7 + 2 * m];
    caw[m] = (const float*)d_in[14 + 2 * m]; cab[m] = (const float*)d_in[15 + 2 * m];
  }
  const float* ffw1 = (const float*)d_in[22]; const float* ffb1 = (const float*)d_in[23];
  const float* ffw2 = (const float*)d_in[24]; const float* ffb2 = (const float*)d_in[25];
  const float* lng[3] = {(const float*)d_in[26], (const float*)d_in[28], (const float*)d_in[30]};
  const float* lnb[3] = {(const float*)d_in[27], (const float*)d_in[29], (const float*)d_in[31]};
  const float* fcw = (const float*)d_in[32]; const float* fcb = (const float*)d_in[33];

  // workspace carve (256B aligned)
  char* ws = (char*)d_ws;
  size_t off = 0;
  auto carve = [&](size_t bytes) -> void* {
    void* p = ws + off; off = (off + bytes + 255) & ~(size_t)255; return p;
  };
  bf16* wsa_bf[4]; bf16* wca_bf[4];
  for (int m = 0; m < 4; ++m) wsa_bf[m] = (bf16*)carve((size_t)NLB * DB * DB * 2);
  for (int m = 0; m < 4; ++m) wca_bf[m] = (bf16*)carve((size_t)NLB * DB * DB * 2);
  bf16* wff1_bf = (bf16*)carve((size_t)NLB * DB * FB * 2);
  bf16* wff2_bf = (bf16*)carve((size_t)NLB * FB * DB * 2);
  bf16* wfc_bf  = (bf16*)carve((size_t)DB * VB * 2);
  bf16* encb    = (bf16*)carve((size_t)Mk * DB * 2);
  float* x      = (float*)carve((size_t)Mq * DB * 4);
  bf16* xb      = (bf16*)carve((size_t)Mq * DB * 2);
  bf16* qb      = (bf16*)carve((size_t)Mq * DB * 2);
  bf16* kb      = (bf16*)carve((size_t)Mq * DB * 2);
  bf16* vb      = (bf16*)carve((size_t)Mq * DB * 2);
  bf16* ctxb    = (bf16*)carve((size_t)Mq * DB * 2);
  float* tmp    = (float*)carve((size_t)Mq * DB * 4);
  bf16* ffh     = (bf16*)carve((size_t)Mq * FB * 2);

  float* logits = (float*)d_out;
  float* attn   = logits + (size_t)BQ * LCB * VB;

  auto cvt = [&](const float* src, bf16* dst, size_t n) {
    const int n4 = (int)(n / 4);
    f2bf_kernel<<<(n4 + 255) / 256, 256, 0, stream>>>(src, dst, n4);
  };
  auto gemm = [&](const bf16* A, const bf16* W, const float* bias,
                  float* oF, bf16* oB, int M, int N, int K, int relu) {
    dim3 grid((N + 127) / 128, M / 128);
    if (N % 128 == 0)
      gemm_bf16_kernel<false><<<grid, 256, 0, stream>>>(A, W, bias, oF, oB, M, N, K, relu);
    else
      gemm_bf16_kernel<true><<<grid, 256, 0, stream>>>(A, W, bias, oF, oB, M, N, K, relu);
  };

  // weight conversion (once per launch)
  for (int m = 0; m < 4; ++m) {
    cvt(saw[m], wsa_bf[m], (size_t)NLB * DB * DB);
    cvt(caw[m], wca_bf[m], (size_t)NLB * DB * DB);
  }
  cvt(ffw1, wff1_bf, (size_t)NLB * DB * FB);
  cvt(ffw2, wff2_bf, (size_t)NLB * FB * DB);
  cvt(fcw,  wfc_bf,  (size_t)DB * VB);
  cvt(enc,  encb,    (size_t)Mk * DB);

  embed_kernel<<<Mq, 128, 0, stream>>>(cds, tok_emb, pos_emb, x, xb);

  const dim3 fgrid(LCB / 64, BQ * HB);
  for (int i = 0; i < NLB; ++i) {
    const size_t wDD = (size_t)i * DB * DB;
    // ---- self-attention ----
    gemm(xb, wsa_bf[0] + wDD, sab[0] + i * DB, nullptr, qb, Mq, DB, DB, 0);
    gemm(xb, wsa_bf[1] + wDD, sab[1] + i * DB, nullptr, kb, Mq, DB, DB, 0);
    gemm(xb, wsa_bf[2] + wDD, sab[2] + i * DB, nullptr, vb, Mq, DB, DB, 0);
    flash_attn_kernel<<<fgrid, 128, 0, stream>>>(qb, kb, vb, ctxb, LCB, LCB, 1);
    gemm(ctxb, wsa_bf[3] + wDD, sab[3] + i * DB, tmp, nullptr, Mq, DB, DB, 0);
    add_ln_kernel<<<Mq, 256, 0, stream>>>(x, tmp, lng[0] + i * DB, lnb[0] + i * DB, xb);
    // ---- cross-attention ----
    gemm(xb,   wca_bf[0] + wDD, cab[0] + i * DB, nullptr, qb, Mq, DB, DB, 0);
    gemm(encb, wca_bf[1] + wDD, cab[1] + i * DB, nullptr, kb, Mk, DB, DB, 0);
    gemm(encb, wca_bf[2] + wDD, cab[2] + i * DB, nullptr, vb, Mk, DB, DB, 0);
    flash_attn_kernel<<<fgrid, 128, 0, stream>>>(qb, kb, vb, ctxb, LCB, LPB, 0);
    if (i == NLB - 1)
      ca_probs_kernel<<<dim3(LCB, BQ * HB), 64, 0, stream>>>(qb, kb, attn);
    gemm(ctxb, wca_bf[3] + wDD, cab[3] + i * DB, tmp, nullptr, Mq, DB, DB, 0);
    add_ln_kernel<<<Mq, 256, 0, stream>>>(x, tmp, lng[1] + i * DB, lnb[1] + i * DB, xb);
    // ---- feed-forward ----
    gemm(xb,  wff1_bf + (size_t)i * DB * FB, ffb1 + i * FB, nullptr, ffh, Mq, FB, DB, 1);
    gemm(ffh, wff2_bf + (size_t)i * FB * DB, ffb2 + i * DB, tmp, nullptr, Mq, DB, FB, 0);
    add_ln_kernel<<<Mq, 256, 0, stream>>>(x, tmp, lng[2] + i * DB, lnb[2] + i * DB, xb);
  }
  // ---- output projection (V=70, bounds-checked N) ----
  gemm(xb, wfc_bf, fcb, logits, nullptr, Mq, VB, DB, 0);
}
